// Genesis_49306224558605
// MI455X (gfx1250) — matrix operand order repaired and checked by reading, not compiled, not run on hardware
//
#include <hip/hip_runtime.h>
#include <hip/hip_bf16.h>

// ---------------- problem constants ----------------
constexpr int B_ = 2, S_ = 1024, D_ = 1024, H_ = 16, V_ = 32000, P_ = 1024;
constexpr int NA_ = 4, NP_ = 4, ND_ = 2, NS_ = 2;
constexpr int M_ROWS = B_ * S_;         // 2048 rows for all (B*S, D) GEMMs

// ---------------- vector types ----------------
typedef __attribute__((ext_vector_type(16))) _Float16     v16h;
typedef __attribute__((ext_vector_type(8)))  float        v8f;
typedef __attribute__((ext_vector_type(4)))  unsigned int u32x4;
typedef __attribute__((ext_vector_type(4)))  float        f32x4;

union HFrag { v16h v; u32x4 u[2]; };

// =====================================================================
// Generic fp32 GEMM: C[M,N] = A[M,K] @ W[K,N] (+bias), f16 WMMA inside.
// Optional device-side weight selection: W += sel[0]*selStride.
// Tiles: block = 128x128, 8 waves in 2x4 grid, wave = 64x32 (4x2 WMMA).
// =====================================================================
__global__ __launch_bounds__(256) void gemm_wmma_kernel(
    const float* __restrict__ A, const float* __restrict__ Wt,
    const float* __restrict__ bias, float* __restrict__ C,
    int M, int N, int K, const int* __restrict__ sel, long long selStride)
{
  if (sel) Wt += (size_t)sel[0] * (size_t)selStride;

  __shared__ __align__(16) _Float16 lA[128 * 40]; // [row][k] k-contig, pad 40
  __shared__ __align__(16) _Float16 lB[128 * 40]; // [n][k]   k-contig, pad 40

  const int tid  = threadIdx.x;
  const int lane = tid & 31;
  const int wid  = tid >> 5;
  const int wm   = wid >> 2;          // 0..1  -> 64-row slab
  const int wn   = wid & 3;           // 0..3  -> 32-col slab
  const int m0   = blockIdx.y * 128;
  const int n0   = blockIdx.x * 128;

  v8f acc[4][2];
  #pragma unroll
  for (int i = 0; i < 4; ++i)
    #pragma unroll
    for (int j = 0; j < 2; ++j) acc[i][j] = {};

  const int arow  = tid >> 1;         // 0..127
  const int ahalf = tid & 1;          // 0..1  (16-wide K halves)

  for (int kk = 0; kk < K; kk += 32) {
    // ---- stage A tile (128x32 fp32 -> f16) ----
    {
      const float* src = A + (size_t)(m0 + arow) * K + kk + ahalf * 16;
      _Float16* dst = &lA[arow * 40 + ahalf * 16];
      #pragma unroll
      for (int j = 0; j < 4; ++j) {
        f32x4 v = *reinterpret_cast<const f32x4*>(src + j * 4);
        dst[j * 4 + 0] = (_Float16)v[0];
        dst[j * 4 + 1] = (_Float16)v[1];
        dst[j * 4 + 2] = (_Float16)v[2];
        dst[j * 4 + 3] = (_Float16)v[3];
      }
    }
    // ---- stage B tile transposed (32x128 fp32 -> lB[n][k] f16) ----
    {
      const int n = arow, kh = ahalf;
      #pragma unroll
      for (int kc = 0; kc < 16; ++kc) {
        float v = Wt[(size_t)(kk + kh * 16 + kc) * N + n0 + n];
        lB[n * 40 + kh * 16 + kc] = (_Float16)v;
      }
    }
    __syncthreads();

    const int khalf = lane >> 4;      // lane half selects K sub-ranges
    const int l15   = lane & 15;

    HFrag bfr[2];
    #pragma unroll
    for (int ni = 0; ni < 2; ++ni) {  // B: K 0-7/16-23 (lo lanes), 8-15/24-31 (hi)
      const int n_loc = wn * 32 + ni * 16 + l15;
      const _Float16* p = &lB[n_loc * 40 + khalf * 8];
      bfr[ni].u[0] = *reinterpret_cast<const u32x4*>(p);
      bfr[ni].u[1] = *reinterpret_cast<const u32x4*>(p + 16);
    }
    #pragma unroll
    for (int mi = 0; mi < 4; ++mi) {  // A: K 0-7/16-23 (lo lanes), 8-15/24-31 (hi)
      const int r = wm * 64 + mi * 16 + l15;
      const _Float16* p = &lA[r * 40];
      HFrag afr;
      afr.u[0] = *reinterpret_cast<const u32x4*>(p + khalf * 8);
      afr.u[1] = *reinterpret_cast<const u32x4*>(p + 16 + khalf * 8);
      #pragma unroll
      for (int ni = 0; ni < 2; ++ni) {
        acc[mi][ni] = __builtin_amdgcn_wmma_f32_16x16x32_f16(
            false, afr.v, false, bfr[ni].v, (short)0, acc[mi][ni], false, false);
      }
    }
    __syncthreads();
  }

  // ---- store C (D layout: vgpr r -> M=r + 8*(lane>=16)) ----
  const int l15 = lane & 15, lh = lane >> 4;
  #pragma unroll
  for (int mi = 0; mi < 4; ++mi) {
    #pragma unroll
    for (int ni = 0; ni < 2; ++ni) {
      const int nc = n0 + wn * 32 + ni * 16 + l15;
      const float badd = bias ? bias[nc] : 0.0f;
      #pragma unroll
      for (int r = 0; r < 8; ++r) {
        const int mr = m0 + wm * 64 + mi * 16 + lh * 8 + r;
        C[(size_t)mr * N + nc] = acc[mi][ni][r] + badd;
      }
    }
  }
}

// =====================================================================
// Row LayerNorm + gain (+bias) (+SiLU) (+residual). One block per row.
// =====================================================================
__global__ __launch_bounds__(256) void ln_act_kernel(
    const float* __restrict__ x, const float* __restrict__ g,
    const float* __restrict__ bb, const float* __restrict__ res,
    float* __restrict__ out, int flags,
    const int* __restrict__ sel, long long selStride)
{
  __shared__ float s1[256], s2[256];
  const int row = blockIdx.x, tid = threadIdx.x;
  const long long off = sel ? (long long)sel[0] * selStride : 0;
  const float* gp = g + off;
  const float* bp = bb ? bb + off : nullptr;
  const size_t base = (size_t)row * D_;

  float v[4]; float sum = 0.f, sq = 0.f;
  #pragma unroll
  for (int t = 0; t < 4; ++t) {
    v[t] = x[base + tid + t * 256];
    sum += v[t]; sq += v[t] * v[t];
  }
  s1[tid] = sum; s2[tid] = sq; __syncthreads();
  for (int st = 128; st > 0; st >>= 1) {
    if (tid < st) { s1[tid] += s1[tid + st]; s2[tid] += s2[tid + st]; }
    __syncthreads();
  }
  const float mean = s1[0] * (1.0f / D_);
  const float var  = s2[0] * (1.0f / D_) - mean * mean;
  const float rstd = rsqrtf(var + 1e-5f);
  #pragma unroll
  for (int t = 0; t < 4; ++t) {
    const int d = tid + t * 256;
    float y = (v[t] - mean) * rstd * gp[d];
    if (bp) y += bp[d];
    if (flags & 1) y = y / (1.0f + __expf(-y));   // SiLU
    if (res) y += res[base + d];
    out[base + d] = y;
  }
}

// =====================================================================
// Causal softmax attention, one wave per (b,h,q). hd=64 -> 2 elems/lane.
// =====================================================================
__global__ __launch_bounds__(256) void attention_kernel(
    const float* __restrict__ Q, const float* __restrict__ Kb,
    const float* __restrict__ Vb, float* __restrict__ O)
{
  const int tid  = threadIdx.x;
  const int lane = tid & 31;
  const int w    = blockIdx.x * (256 >> 5) + (tid >> 5);
  const int q = w % S_;
  const int h = (w / S_) % H_;
  const int b = w / (S_ * H_);
  if (b >= B_) return;

  const size_t base = (size_t)(b * S_ + q) * D_ + h * 64;
  const float q0 = Q[base + lane], q1 = Q[base + lane + 32];
  float m = -1e30f, l = 0.f, o0 = 0.f, o1 = 0.f;

  for (int kp = 0; kp <= q; ++kp) {
    const size_t kb = (size_t)(b * S_ + kp) * D_ + h * 64;
    float part = q0 * Kb[kb + lane] + q1 * Kb[kb + lane + 32];
    #pragma unroll
    for (int off = 16; off > 0; off >>= 1) part += __shfl_xor(part, off, 32);
    const float score = part * 0.125f;               // 1/sqrt(64)
    const float mn = fmaxf(m, score);
    const float sc = __expf(m - mn);
    const float p  = __expf(score - mn);
    l  = l * sc + p;
    o0 = o0 * sc + p * Vb[kb + lane];
    o1 = o1 * sc + p * Vb[kb + lane + 32];
    m = mn;
  }
  O[base + lane]      = o0 / l;
  O[base + lane + 32] = o1 / l;
}

// ---------------- small helper kernels ----------------
__global__ void embed_kernel(const int* __restrict__ ids,
                             const float* __restrict__ wte,
                             const float* __restrict__ wpe,
                             float* __restrict__ out)
{
  const size_t i = (size_t)blockIdx.x * blockDim.x + threadIdx.x;
  if (i >= (size_t)B_ * S_ * D_) return;
  const int d = i % D_;
  const int s = (i / D_) % S_;
  const int b = i / ((size_t)D_ * S_);
  const int id = ids[b * S_ + s];
  out[i] = wte[(size_t)id * D_ + d] + wpe[(size_t)s * D_ + d];
}

__global__ void rowmean_kernel(const float* __restrict__ x, float* __restrict__ hm)
{
  const int i = blockIdx.x * blockDim.x + threadIdx.x;   // over B*D
  if (i >= B_ * D_) return;
  const int b = i / D_, d = i % D_;
  float s = 0.f;
  for (int sx = 0; sx < S_; ++sx) s += x[(size_t)(b * S_ + sx) * D_ + d];
  hm[i] = s * (1.0f / S_);
}

__global__ __launch_bounds__(256) void volt_kernel(
    const float* __restrict__ hm, const float* __restrict__ soma,
    float* __restrict__ volt)
{
  __shared__ float sm[256];
  const int b = blockIdx.x, tid = threadIdx.x;
  float s = 0.f;
  #pragma unroll
  for (int t = 0; t < 4; ++t) { const int d = tid + t * 256; s += hm[b * D_ + d] * soma[d]; }
  sm[tid] = s; __syncthreads();
  for (int st = 128; st > 0; st >>= 1) { if (tid < st) sm[tid] += sm[tid + st]; __syncthreads(); }
  if (tid == 0) volt[b] = 1.0f / (1.0f + __expf(-sm[0]));
}

__global__ void argmax_kernel(const float* __restrict__ volt, int n, int nb,
                              int* __restrict__ outIdx)
{
  if (threadIdx.x == 0 && blockIdx.x == 0) {
    int best = 0; float bv = -1e30f;
    for (int i = 0; i < n; ++i) {
      float m = 0.f;
      for (int b = 0; b < nb; ++b) m += volt[i * nb + b];
      m /= nb;
      if (m > bv) { bv = m; best = i; }
    }
    outIdx[0] = best;
  }
}

__global__ void mask_kernel(const float* __restrict__ volt,
                            const float* __restrict__ thr,
                            const int* __restrict__ j, float* __restrict__ maskF)
{
  const int b = threadIdx.x;
  if (b >= B_) return;
  const int jj = j[0];
  maskF[b] = (volt[jj * B_ + b] > thr[jj]) ? 1.0f : 0.0f;
}

__global__ void apply_mask_kernel(const float* __restrict__ h,
                                  const float* __restrict__ maskF,
                                  float* __restrict__ out)
{
  const size_t i = (size_t)blockIdx.x * blockDim.x + threadIdx.x;
  if (i >= (size_t)B_ * S_ * D_) return;
  const int b = i / ((size_t)S_ * D_);
  out[i] = maskF[b] * h[i];
}

__global__ void combine_kernel(float* __restrict__ h,
                               const float* __restrict__ m0,
                               const float* __restrict__ proc,
                               const float* __restrict__ maskF)
{
  const size_t i = (size_t)blockIdx.x * blockDim.x + threadIdx.x;
  if (i >= (size_t)B_ * S_ * D_) return;
  const int b = i / ((size_t)S_ * D_);
  if (maskF[b] > 0.5f) h[i] = m0[i] + proc[i];
}

// =====================================================================
// Host-side orchestration
// =====================================================================
extern "C" void kernel_launch(void* const* d_in, const int* in_sizes, int n_in,
                              void* d_out, int out_size, void* d_ws, size_t ws_size,
                              hipStream_t stream)
{
  (void)in_sizes; (void)n_in; (void)out_size; (void)ws_size;

  // ---- inputs (setup_inputs order) ----
  const int*   input_ids = (const int*)  d_in[0];
  const float* wte   = (const float*)d_in[1];
  const float* wpe   = (const float*)d_in[2];
  const float* r_w   = (const float*)d_in[3];
  const float* r_b   = (const float*)d_in[4];
  const float* r_g   = (const float*)d_in[5];
  const float* r_bb  = (const float*)d_in[6];
  const float* a_dend_w = (const float*)d_in[7];
  const float* a_dend_g = (const float*)d_in[8];
  const float* a_soma_w = (const float*)d_in[9];
  const float* a_qw  = (const float*)d_in[10];
  const float* a_kw  = (const float*)d_in[11];
  const float* a_vw  = (const float*)d_in[12];
  const float* a_ow  = (const float*)d_in[13];
  const float* a_axon_g = (const float*)d_in[14];
  const float* a_syn_w  = (const float*)d_in[15];
  const float* a_syn_g  = (const float*)d_in[16];
  const float* a_nln_g  = (const float*)d_in[17];
  const float* p_dend_w = (const float*)d_in[18];
  const float* p_dend_g = (const float*)d_in[19];
  const float* p_soma_w = (const float*)d_in[20];
  const float* p_thr    = (const float*)d_in[21];
  const float* p_ow     = (const float*)d_in[22];
  const float* p_axon_g = (const float*)d_in[23];
  const float* p_syn_w  = (const float*)d_in[24];
  const float* p_syn_g  = (const float*)d_in[25];
  const float* c_w  = (const float*)d_in[26];
  const float* c_b  = (const float*)d_in[27];
  const float* c_g  = (const float*)d_in[28];
  const float* c_bb = (const float*)d_in[29];
  const float* b_w  = (const float*)d_in[30];
  const float* b_b  = (const float*)d_in[31];
  const float* b_g  = (const float*)d_in[32];
  const float* b_bb = (const float*)d_in[33];
  const float* lm_w = (const float*)d_in[34];
  const float* lm_b = (const float*)d_in[35];
  float* outp = (float*)d_out;

  // ---- workspace layout ----
  float* ws = (float*)d_ws;
  const size_t BSD = (size_t)B_ * S_ * D_;
  float* H0 = ws + 0 * BSD;
  float* T0 = ws + 1 * BSD;
  float* T1 = ws + 2 * BSD;
  float* Qb = ws + 3 * BSD;
  float* Kb = ws + 4 * BSD;
  float* Vv = ws + 5 * BSD;
  float* Ob = ws + 6 * BSD;
  float* N0 = ws + 7 * BSD;
  float* M0 = ws + 8 * BSD;
  float* HM    = ws + 9 * BSD;         // B*D
  float* VOLT  = HM + (size_t)B_ * D_; // up to NP*B
  float* MASKF = VOLT + 16;
  int* IDXA = (int*)(MASKF + 4);
  int* IDXP = IDXA + 1;

  const dim3 blk(256);
  const int EW = (int)((BSD + 255) / 256);     // elementwise grid
  const long long DD = (long long)D_ * D_;

  auto gemm = [&](const float* A, const float* W, const float* bias, float* C,
                  int N, const int* sel, long long ss) {
    dim3 g(N / 128, M_ROWS / 128);
    gemm_wmma_kernel<<<g, blk, 0, stream>>>(A, W, bias, C, M_ROWS, N, D_, sel, ss);
  };
  auto ln = [&](const float* x, const float* g, const float* bb, const float* res,
                float* out, int flags, const int* sel, long long ss) {
    ln_act_kernel<<<dim3(M_ROWS), blk, 0, stream>>>(x, g, bb, res, out, flags, sel, ss);
  };

  // 1) embed: h = wte[ids] + wpe[pos]
  embed_kernel<<<dim3(EW), blk, 0, stream>>>(input_ids, wte, wpe, H0);

  // 2) h = silu(ln(h @ r_w + r_b, r_g, r_bb))
  gemm(H0, r_w, r_b, T0, D_, nullptr, 0);
  ln(T0, r_g, r_bb, nullptr, H0, 1, nullptr, 0);

  // 3) a-interact -> idx
  for (int n = 0; n < NA_; ++n) {
    gemm(H0, a_dend_w + ((size_t)n * ND_ + 0) * DD, nullptr, T0, D_, nullptr, 0);
    ln(T0, a_dend_g + ((size_t)n * ND_ + 0) * D_, nullptr, nullptr, T1, 1, nullptr, 0);
    gemm(T1, a_dend_w + ((size_t)n * ND_ + 1) * DD, nullptr, T0, D_, nullptr, 0);
    ln(T0, a_dend_g + ((size_t)n * ND_ + 1) * D_, nullptr, nullptr, T1, 1, nullptr, 0);
    rowmean_kernel<<<dim3((B_ * D_ + 255) / 256), blk, 0, stream>>>(T1, HM);
    volt_kernel<<<dim3(B_), blk, 0, stream>>>(HM, a_soma_w + (size_t)n * D_, VOLT + n * B_);
  }
  argmax_kernel<<<1, 1, 0, stream>>>(VOLT, NA_, B_, IDXA);

  // 4) attention with expert idx (device-selected weights)
  gemm(H0, a_qw, nullptr, Qb, D_, IDXA, DD);
  gemm(H0, a_kw, nullptr, Kb, D_, IDXA, DD);
  gemm(H0, a_vw, nullptr, Vv, D_, IDXA, DD);
  attention_kernel<<<dim3(B_ * H_ * S_ / 8), blk, 0, stream>>>(Qb, Kb, Vv, Ob);

  // 5) h = silu(ln(o @ a_ow[idx], a_axon_g[idx])) + h
  gemm(Ob, a_ow, nullptr, T0, D_, IDXA, DD);
  ln(T0, a_axon_g, nullptr, H0, H0, 1, IDXA, D_);

  // 6) synapse stack + residual (res = H0, kept intact until final add)
  gemm(H0, a_syn_w + 0 * DD, nullptr, T0, D_, IDXA, (long long)NS_ * DD);
  ln(T0, a_syn_g + 0 * D_, nullptr, nullptr, T1, 1, IDXA, (long long)NS_ * D_);
  gemm(T1, a_syn_w + 1 * DD, nullptr, T0, D_, IDXA, (long long)NS_ * DD);
  ln(T0, a_syn_g + 1 * D_, nullptr, H0, H0, 1, IDXA, (long long)NS_ * D_);

  // 7) p-interact -> j, volt
  for (int n = 0; n < NP_; ++n) {
    gemm(H0, p_dend_w + ((size_t)n * ND_ + 0) * DD, nullptr, T0, D_, nullptr, 0);
    ln(T0, p_dend_g + ((size_t)n * ND_ + 0) * D_, nullptr, nullptr, T1, 1, nullptr, 0);
    gemm(T1, p_dend_w + ((size_t)n * ND_ + 1) * DD, nullptr, T0, D_, nullptr, 0);
    ln(T0, p_dend_g + ((size_t)n * ND_ + 1) * D_, nullptr, nullptr, T1, 1, nullptr, 0);
    rowmean_kernel<<<dim3((B_ * D_ + 255) / 256), blk, 0, stream>>>(T1, HM);
    volt_kernel<<<dim3(B_), blk, 0, stream>>>(HM, p_soma_w + (size_t)n * D_, VOLT + n * B_);
  }
  argmax_kernel<<<1, 1, 0, stream>>>(VOLT, NP_, B_, IDXP);
  mask_kernel<<<1, 32, 0, stream>>>(VOLT, p_thr, IDXP, MASKF);

  // 8) masked path
  apply_mask_kernel<<<dim3(EW), blk, 0, stream>>>(H0, MASKF, M0);
  gemm(M0, p_ow, nullptr, T0, D_, IDXP, DD);
  ln(T0, p_axon_g, nullptr, M0, N0, 1, IDXP, D_);
  gemm(N0, p_syn_w + 0 * DD, nullptr, T0, D_, IDXP, (long long)NS_ * DD);
  ln(T0, p_syn_g + 0 * D_, nullptr, nullptr, T1, 1, IDXP, (long long)NS_ * D_);
  gemm(T1, p_syn_w + 1 * DD, nullptr, T0, D_, IDXP, (long long)NS_ * DD);
  ln(T0, p_syn_g + 1 * D_, nullptr, N0, N0, 1, IDXP, (long long)NS_ * D_);

  // 9) processed = ln(nh, a_nln_g[idx]); h = mask ? masked + processed : h
  ln(N0, a_nln_g, nullptr, nullptr, T1, 0, IDXA, D_);
  combine_kernel<<<dim3(EW), blk, 0, stream>>>(H0, M0, T1, MASKF);

  // 10) c / b layers
  gemm(H0, c_w, c_b, T0, D_, nullptr, 0);
  ln(T0, c_g, c_bb, nullptr, H0, 1, nullptr, 0);
  gemm(H0, b_w, b_b, T0, D_, nullptr, 0);
  ln(T0, b_g, b_bb, nullptr, H0, 0, nullptr, 0);

  // 11) LM head: (2048 x 1024) @ (1024 x 32000) + lm_b -> d_out
  gemm(H0, lm_w, lm_b, outp, V_, nullptr, 0);
}
